// DynamicConv1dTBC_60576218743246
// MI455X (gfx1250) — compile-verified, matrix operand order checked by reading, not run on hardware
//
#include <hip/hip_runtime.h>

typedef unsigned short u16;
typedef unsigned int   u32;
typedef __attribute__((ext_vector_type(16))) __bf16 v16bf;
typedef __attribute__((ext_vector_type(8)))  float  v8f;
typedef __attribute__((ext_vector_type(4)))  unsigned int   uint4v;
typedef __attribute__((ext_vector_type(4)))  unsigned short us4v;
typedef __attribute__((ext_vector_type(4)))  float          f4v;
typedef int i4_t __attribute__((__vector_size__(16)));   // matches builtin param type

union Frag16 { v16bf v; uint4v q[2]; };

// ---- scalar helpers (bit-exact, no reliance on __bf16 scalar codegen) ----
static __device__ __forceinline__ u16 f2bf(float f) {
    u32 x; __builtin_memcpy(&x, &f, 4);
    u32 r = x + 0x7fffu + ((x >> 16) & 1u);   // round-to-nearest-even
    return (u16)(r >> 16);
}
static __device__ __forceinline__ float bf2f(u16 u) {
    u32 x = ((u32)u) << 16;
    float f; __builtin_memcpy(&f, &x, 4);
    return f;
}

// ---- CDNA5 async global->LDS path (ASYNCcnt), with safe fallback ----
#if defined(__AMDGCN__) && __has_builtin(__builtin_amdgcn_global_load_async_to_lds_b128)
#define USE_ASYNC 1
#else
#define USE_ASYNC 0
#endif

#if USE_ASYNC
static __device__ __forceinline__ void async_cp16(const u16* g, u16* l) {
    __builtin_amdgcn_global_load_async_to_lds_b128(
        (__attribute__((address_space(1))) i4_t*)g,
        (__attribute__((address_space(3))) i4_t*)l,
        0, 0);
}
static __device__ __forceinline__ void wait_async0() {
#if __has_builtin(__builtin_amdgcn_s_wait_asynccnt)
    __builtin_amdgcn_s_wait_asynccnt(0);
#else
    asm volatile("s_wait_asynccnt 0x0" ::: "memory");
#endif
}
#endif

// ---------------- problem constants ----------------
#define TT   1024
#define BB   16
#define CC   1024
#define HH   16
#define KK   7
#define PADL 3
#define OUTC 1024
#define MROWS (TT * BB)          // 16384
#define NPAD  128                // padded H*K

// ---------------- conversion kernels ----------------
__global__ void cvt_f32_to_bf16_k(const float* __restrict__ in, u16* __restrict__ out, int n4) {
    int i = blockIdx.x * blockDim.x + threadIdx.x;
    if (i >= n4) return;
    f4v v = ((const f4v*)in)[i];
    us4v o;
    o.x = f2bf(v.x); o.y = f2bf(v.y); o.z = f2bf(v.z); o.w = f2bf(v.w);
    ((us4v*)out)[i] = o;
}

// in: f32 [R][Cc] row-major  ->  out: bf16 [Cc][R]
__global__ void transpose_to_bf16_k(const float* __restrict__ in, u16* __restrict__ out, int R, int Cc) {
    __shared__ float tile[16][17];
    int c0 = blockIdx.x * 16, r0 = blockIdx.y * 16;
    tile[threadIdx.y][threadIdx.x] = in[(size_t)(r0 + threadIdx.y) * Cc + (c0 + threadIdx.x)];
    __syncthreads();
    out[(size_t)(c0 + threadIdx.y) * R + (r0 + threadIdx.x)] = f2bf(tile[threadIdx.x][threadIdx.y]);
}

// Wq [1024][112] f32 -> Wqt_pad [128][1024] bf16 (rows >= 112 zero); bq -> bq_pad[128]
__global__ void prep_wq_k(const float* __restrict__ Wq, const float* __restrict__ bq,
                          u16* __restrict__ Wqt, float* __restrict__ bqp) {
    int idx = blockIdx.x * 256 + threadIdx.x;       // 0 .. 131071
    int n = idx >> 10, k = idx & 1023;
    float v = (n < HH * KK) ? Wq[(size_t)k * (HH * KK) + n] : 0.f;
    Wqt[idx] = f2bf(v);
    if (blockIdx.x == 0 && threadIdx.x < NPAD)
        bqp[threadIdx.x] = (threadIdx.x < HH * KK) ? bq[threadIdx.x] : 0.f;
}

// ---------------- WMMA GEMM: C = A(bf16 [M][K]) * Bt(bf16 [N][K])^T + bias ----------------
#define BM 128
#define BN 128
#define BKt 64   // two WMMA K-steps per LDS buffer -> half the barriers
#define LDT 72   // LDS row stride (elems): 64 + 8 pad, keeps 16B alignment, kills conflicts

__global__ __launch_bounds__(256) void gemm_bf16_wmma_k(
    const u16* __restrict__ A, const u16* __restrict__ Bt,
    const float* __restrict__ bias,
    float* __restrict__ outF, u16* __restrict__ outH,
    int M, int N, int K)
{
    __shared__ __align__(16) u16 lsA[2][BM * LDT];   // 36 KB
    __shared__ __align__(16) u16 lsB[2][BM * LDT];   // 36 KB

    const int tid   = threadIdx.x;
    const int wave  = tid >> 5;
    const int lane  = tid & 31;
    const int waveM = wave >> 2;                 // 0..1
    const int waveN = wave & 3;                  // 0..3
    const int blockM = blockIdx.y * BM;
    const int blockN = blockIdx.x * BN;

    // global->LDS tile mapping: 256 threads cover 128 rows x 64 cols, 4x16B chunks each
    const int lrow = tid >> 1;                   // 0..127
    const int lcol = (tid & 1) * 32;             // 0 or 32 (elems)
    const u16* gA = A  + (size_t)(blockM + lrow) * K + lcol;
    const u16* gB = Bt + (size_t)(blockN + lrow) * K + lcol;
    const int sOff = lrow * LDT + lcol;

    // prologue: tile 0 -> LDS[0]
#if USE_ASYNC
    #pragma unroll
    for (int c = 0; c < 4; ++c) {
        async_cp16(gA + c * 8, &lsA[0][sOff + c * 8]);
        async_cp16(gB + c * 8, &lsB[0][sOff + c * 8]);
    }
    wait_async0();
#else
    {
        uint4v ra[4], rb[4];
        #pragma unroll
        for (int c = 0; c < 4; ++c) { ra[c] = *(const uint4v*)(gA + c * 8); rb[c] = *(const uint4v*)(gB + c * 8); }
        #pragma unroll
        for (int c = 0; c < 4; ++c) {
            *(uint4v*)&lsA[0][sOff + c * 8] = ra[c];
            *(uint4v*)&lsB[0][sOff + c * 8] = rb[c];
        }
    }
#endif
    __syncthreads();

    v8f acc[4][2] = {};

    // ISA 16-bit A/B fragment addressing (16x32, wave32)
    const int fm = lane & 15;
    const int fk = (lane >> 4) * 8;              // 0 or 8 elems

    const int NT = K / BKt;
    int buf = 0;
    for (int kt = 0; kt < NT; ++kt) {
        const bool more = (kt + 1 < NT);
#if USE_ASYNC
        if (more) {
            const u16* gA2 = gA + (size_t)(kt + 1) * BKt;
            const u16* gB2 = gB + (size_t)(kt + 1) * BKt;
            #pragma unroll
            for (int c = 0; c < 4; ++c) {
                async_cp16(gA2 + c * 8, &lsA[buf ^ 1][sOff + c * 8]);
                async_cp16(gB2 + c * 8, &lsB[buf ^ 1][sOff + c * 8]);
            }
        }
#else
        uint4v ra[4], rb[4];
        if (more) {
            const u16* gA2 = gA + (size_t)(kt + 1) * BKt;
            const u16* gB2 = gB + (size_t)(kt + 1) * BKt;
            #pragma unroll
            for (int c = 0; c < 4; ++c) { ra[c] = *(const uint4v*)(gA2 + c * 8); rb[c] = *(const uint4v*)(gB2 + c * 8); }
        }
#endif
        // two WMMA K-steps out of the current LDS buffer
        #pragma unroll
        for (int ks = 0; ks < 2; ++ks) {
            const int kk = ks * 32;
            Frag16 a[4], b[2];
            #pragma unroll
            for (int i = 0; i < 4; ++i) {
                const u16* p = &lsA[buf][(waveM * 64 + i * 16 + fm) * LDT + kk + fk];
                a[i].q[0] = *(const uint4v*)p;
                a[i].q[1] = *(const uint4v*)(p + 16);
            }
            #pragma unroll
            for (int j = 0; j < 2; ++j) {
                const u16* p = &lsB[buf][(waveN * 32 + j * 16 + fm) * LDT + kk + fk];
                b[j].q[0] = *(const uint4v*)p;
                b[j].q[1] = *(const uint4v*)(p + 16);
            }
            #pragma unroll
            for (int i = 0; i < 4; ++i)
                #pragma unroll
                for (int j = 0; j < 2; ++j)
                    acc[i][j] = __builtin_amdgcn_wmma_f32_16x16x32_bf16(
                        false, a[i].v, false, b[j].v, (short)0, acc[i][j], false, false);
        }
#if USE_ASYNC
        wait_async0();
#else
        if (more) {
            #pragma unroll
            for (int c = 0; c < 4; ++c) {
                *(uint4v*)&lsA[buf ^ 1][sOff + c * 8] = ra[c];
                *(uint4v*)&lsB[buf ^ 1][sOff + c * 8] = rb[c];
            }
        }
#endif
        __syncthreads();
        buf ^= 1;
    }

    // epilogue: C/D layout -> lane 0-15: M=r, lane 16-31: M=r+8; N = lane&15
    const int rbase = blockM + waveM * 64 + ((lane >> 4) * 8);
    const int cbase = blockN + waveN * 32 + (lane & 15);
    #pragma unroll
    for (int i = 0; i < 4; ++i) {
        #pragma unroll
        for (int j = 0; j < 2; ++j) {
            int col = cbase + j * 16;
            float bv = bias[col];
            #pragma unroll
            for (int r = 0; r < 8; ++r) {
                int row = rbase + i * 16 + r;
                float v = acc[i][j][r] + bv;
                if (outF) outF[(size_t)row * N + col] = v;
                if (outH) outH[(size_t)row * N + col] = f2bf(v);
            }
        }
    }
}

// ---------------- softmax over K=7 taps per (row, head) ----------------
__global__ void softmax7_k(const float* __restrict__ logits /*[MROWS][128]*/,
                           float* __restrict__ w /*[MROWS][112]*/) {
    int idx = blockIdx.x * 256 + threadIdx.x;       // row*16 + head
    if (idx >= MROWS * HH) return;
    int row = idx >> 4, head = idx & 15;
    const float* p = logits + (size_t)row * NPAD + head * KK;
    float e[KK];
    float m = p[0];
    #pragma unroll
    for (int k = 1; k < KK; ++k) m = fmaxf(m, p[k]);
    float s = 0.f;
    #pragma unroll
    for (int k = 0; k < KK; ++k) { e[k] = __expf(p[k] - m); s += e[k]; }
    float inv = 1.f / s;
    float* o = w + (size_t)row * (HH * KK) + head * KK;
    #pragma unroll
    for (int k = 0; k < KK; ++k) o[k] = e[k] * inv;
}

// ---------------- dynamic conv: out = sum_k h[t+k-3] * w[k] + conv_bias ----------------
__global__ __launch_bounds__(256) void dynconv_k(const u16* __restrict__ h /*[MROWS][CC] bf16*/,
                                                 const float* __restrict__ w /*[MROWS][112]*/,
                                                 const float* __restrict__ cb /*[CC]*/,
                                                 u16* __restrict__ out /*[MROWS][CC] bf16*/) {
    int tb = blockIdx.x;                 // 0..MROWS-1   (= t*B + b)
    int c0 = threadIdx.x * 4;            // 4 channels per thread
    int head = c0 >> 6;                  // CC/HH = 64 channels per head
    float wk[KK];
    #pragma unroll
    for (int k = 0; k < KK; ++k) wk[k] = w[(size_t)tb * (HH * KK) + head * KK + k];
    float a0 = 0.f, a1 = 0.f, a2 = 0.f, a3 = 0.f;
    #pragma unroll
    for (int k = 0; k < KK; ++k) {
        int row = tb + (k - PADL) * BB;  // time shift of (k-3) == row shift of (k-3)*B
        if (row < 0 || row >= MROWS) continue;
        us4v hv = *(const us4v*)(h + (size_t)row * CC + c0);
        a0 += bf2f(hv.x) * wk[k];
        a1 += bf2f(hv.y) * wk[k];
        a2 += bf2f(hv.z) * wk[k];
        a3 += bf2f(hv.w) * wk[k];
    }
    us4v o;
    o.x = f2bf(a0 + cb[c0 + 0]);
    o.y = f2bf(a1 + cb[c0 + 1]);
    o.z = f2bf(a2 + cb[c0 + 2]);
    o.w = f2bf(a3 + cb[c0 + 3]);
    *(us4v*)(out + (size_t)tb * CC + c0) = o;
}

// ---------------- launcher ----------------
extern "C" void kernel_launch(void* const* d_in, const int* in_sizes, int n_in,
                              void* d_out, int out_size, void* d_ws, size_t ws_size,
                              hipStream_t stream) {
    const float* x   = (const float*)d_in[0];
    const float* W1  = (const float*)d_in[1];
    const float* b1  = (const float*)d_in[2];
    const float* Wq  = (const float*)d_in[3];
    const float* bq  = (const float*)d_in[4];
    const float* cb  = (const float*)d_in[5];
    const float* W2  = (const float*)d_in[6];
    const float* b2  = (const float*)d_in[7];
    float* out = (float*)d_out;

    char* ws = (char*)d_ws;
    size_t o = 0;
    auto take = [&](size_t bytes) { char* p = ws + o; o += (bytes + 255) & ~(size_t)255; return p; };
    u16*   xb    = (u16*)  take((size_t)MROWS * CC * 2);      // x bf16
    u16*   W1t   = (u16*)  take((size_t)CC * CC * 2);         // W1^T bf16  [N][K]
    u16*   W2t   = (u16*)  take((size_t)CC * OUTC * 2);       // W2^T bf16  [N][K]
    u16*   Wqt   = (u16*)  take((size_t)NPAD * CC * 2);       // Wq^T padded [128][1024]
    float* bqp   = (float*)take((size_t)NPAD * 4);            // padded bq
    u16*   hb    = (u16*)  take((size_t)MROWS * CC * 2);      // h bf16
    float* logit = (float*)take((size_t)MROWS * NPAD * 4);    // h@Wq+bq (padded)
    float* wsf   = (float*)take((size_t)MROWS * HH * KK * 4); // softmax weights
    u16*   cvb   = (u16*)  take((size_t)MROWS * CC * 2);      // conv output bf16
    (void)o; (void)ws_size; (void)in_sizes; (void)n_in; (void)out_size;

    // 1) conversions / weight prep
    {
        int n4 = MROWS * CC / 4;
        cvt_f32_to_bf16_k<<<dim3((n4 + 255) / 256), dim3(256), 0, stream>>>(x, xb, n4);
    }
    transpose_to_bf16_k<<<dim3(CC / 16, CC / 16), dim3(16, 16), 0, stream>>>(W1, W1t, CC, CC);
    transpose_to_bf16_k<<<dim3(OUTC / 16, CC / 16), dim3(16, 16), 0, stream>>>(W2, W2t, CC, OUTC);
    prep_wq_k<<<dim3(NPAD * CC / 256), dim3(256), 0, stream>>>(Wq, bq, Wqt, bqp);

    // 2) GEMM1: h = x @ W1 + b1 (bf16 out only)
    gemm_bf16_wmma_k<<<dim3(CC / BN, MROWS / BM), dim3(256), 0, stream>>>(
        xb, W1t, b1, nullptr, hb, MROWS, CC, CC);

    // 3) GEMM2: logits = h @ Wq + bq (f32 out, N padded to 128)
    gemm_bf16_wmma_k<<<dim3(NPAD / BN, MROWS / BM), dim3(256), 0, stream>>>(
        hb, Wqt, bqp, logit, nullptr, MROWS, NPAD, CC);

    // 4) softmax over the 7 taps of each head
    softmax7_k<<<dim3((MROWS * HH + 255) / 256), dim3(256), 0, stream>>>(logit, wsf);

    // 5) dynamic conv (+ conv_bias), bf16 out
    dynconv_k<<<dim3(MROWS), dim3(256), 0, stream>>>(hb, wsf, cb, cvb);

    // 6) GEMM3: out = conv @ W2 + b2 (f32 -> d_out)
    gemm_bf16_wmma_k<<<dim3(OUTC / BN, MROWS / BM), dim3(256), 0, stream>>>(
        cvb, W2t, b2, out, nullptr, MROWS, OUTC, CC);
}
